// Fusion_Block1_74534862454881
// MI455X (gfx1250) — matrix operand order repaired and checked by reading, not compiled, not run on hardware
//
#include <hip/hip_runtime.h>
#include <hip/hip_bf16.h>

#define DEVINL __device__ __forceinline__

typedef __bf16 bf16;
typedef bf16 v16bf __attribute__((ext_vector_type(16)));
typedef bf16 v8bf  __attribute__((ext_vector_type(8)));
typedef float v8f  __attribute__((ext_vector_type(8)));
typedef int b128i __attribute__((__vector_size__(16)));   // matches builtin param type

#define Bn  2
#define Tt  8
#define Hh  56
#define Ww  56
#define Np  25088      /* Tt*Hh*Ww */
#define NT  1568       /* Np/16 */
#define CHn 12
#define Mm  588        /* CH*7*7 */
#define Mp  608        /* padded to x32 */
#define CRr 16
#define Cf  256

#if defined(__gfx1250__) && __has_builtin(__builtin_amdgcn_global_load_async_to_lds_b128)
#define ASYNC_LDS_COPY 1
#else
#define ASYNC_LDS_COPY 0
#endif

// ---------------------------------------------------------------------------
// WMMA fragment helpers (bf16 16x16x32, wave32). Layouts per CDNA5 ISA 7.12.2:
//  A (16x32): lane = M row (l&15); lanes<16 hold K 0..7,16..23, lanes>=16 hold
//             K 8..15,24..31  -> two 8-elem contiguous chunks.
//  B (32x16): lane = N col (l&15); same K split as A (lanes<16 K 0..7,16..23,
//             lanes>=16 K 8..15,24..31) -> two 8-elem contiguous chunks.
//  C/D f32:   VGPR r -> row r + (lane<16?0:8), col = lane&15.
// ---------------------------------------------------------------------------
DEVINL v16bf load_a_frag(const bf16* __restrict__ rowptr, int kstep, int lane) {
  const bf16* p = rowptr + kstep * 32 + ((lane & 16) ? 8 : 0);
  v8bf c0 = *(const v8bf*)(p);
  v8bf c1 = *(const v8bf*)(p + 16);
  v16bf a;
#pragma unroll
  for (int i = 0; i < 8; ++i) { a[i] = c0[i]; a[i + 8] = c1[i]; }
  return a;
}

DEVINL v16bf load_b_frag(const bf16* __restrict__ colptr, int kstep, int lane) {
  const bf16* p = colptr + kstep * 32 + ((lane & 16) ? 8 : 0);
  v8bf c0 = *(const v8bf*)(p);
  v8bf c1 = *(const v8bf*)(p + 16);
  v16bf b;
#pragma unroll
  for (int i = 0; i < 8; ++i) { b[i] = c0[i]; b[i + 8] = c1[i]; }
  return b;
}

DEVINL v16bf zero16() {
  v16bf z;
#pragma unroll
  for (int i = 0; i < 16; ++i) z[i] = (bf16)0.0f;
  return z;
}

DEVINL v8f wmma_bf16(v16bf a, v16bf b, v8f c) {
  return __builtin_amdgcn_wmma_f32_16x16x32_bf16(false, a, false, b, (short)0, c,
                                                 false, false);
}

// Cooperative copy of one 16x608 bf16 A-slab (19456 B) into LDS.
// Uses the gfx1250 async global->LDS path when available (ASYNCcnt-tracked).
DEVINL void stage_slab(const bf16* __restrict__ src, bf16* dst, int tid) {
  for (int i = tid; i < (16 * Mp) / 8; i += 128) {
#if ASYNC_LDS_COPY
    __builtin_amdgcn_global_load_async_to_lds_b128(
        (__attribute__((address_space(1))) b128i*)(src + i * 8),
        (__attribute__((address_space(3))) b128i*)(dst + i * 8), 0, 0);
#else
    *(v8bf*)(dst + i * 8) = *(const v8bf*)(src + i * 8);
#endif
  }
}

DEVINL void async_wait_all() {
#if ASYNC_LDS_COPY
#if __has_builtin(__builtin_amdgcn_s_wait_asynccnt)
  __builtin_amdgcn_s_wait_asynccnt(0);
#else
  asm volatile("s_wait_asynccnt 0x0" ::: "memory");
#endif
#endif
}

// ---------------------------------------------------------------------------
// Prep kernels
// ---------------------------------------------------------------------------
__global__ void k_cvt_bf16(const float* __restrict__ s, bf16* __restrict__ d, int n) {
  int i = blockIdx.x * 256 + threadIdx.x;
  if (i < n) d[i] = (bf16)s[i];
}

// W_inv (49,588) -> (64,608) zero-padded row-major bf16
__global__ void k_pack_winv(const float* __restrict__ w, bf16* __restrict__ d) {
  int i = blockIdx.x * 256 + threadIdx.x;
  if (i >= 64 * Mp) return;
  int r = i / Mp, m = i % Mp;
  d[i] = (r < 49 && m < Mm) ? (bf16)w[r * Mm + m] : (bf16)0.0f;
}

// W_att (13,588,5,5,5) -> [tap 125][oc 16][m 608] bf16 (zero-padded)
__global__ void k_pack_watt(const float* __restrict__ w, bf16* __restrict__ d) {
  long long i = (long long)blockIdx.x * 256 + threadIdx.x;
  if (i >= (long long)125 * 16 * Mp) return;
  int m = (int)(i % Mp);
  long long r = i / Mp;
  int oc = (int)(r % 16);
  int tap = (int)(r / 16);
  d[i] = (oc < 13 && m < Mm) ? (bf16)w[((long long)(oc * Mm + m)) * 125 + tap]
                             : (bf16)0.0f;
}

// x_CT (b,64,N) fp32 -> (b,N,64) bf16
__global__ void k_transpose_xct(const float* __restrict__ x, bf16* __restrict__ xt) {
  int n = blockIdx.x * 256 + threadIdx.x;
  int b = blockIdx.y;
  if (n >= Np) return;
  bf16* o = xt + ((size_t)b * Np + n) * 64;
#pragma unroll 4
  for (int c = 0; c < 64; ++c)
    o[c] = (bf16)x[((size_t)(b * 64 + c)) * Np + n];
}

// adaptive pool of x_WSI: P[b][m][ci] = mean over 8x8 block (fd=1)
__global__ void k_pool_wsi(const float* __restrict__ xwsi, float* __restrict__ P) {
  int ci = threadIdx.x;           // 64
  int m = blockIdx.x;             // 588
  int b = blockIdx.y;
  int d = m / 49, kh = (m % 49) / 7, kw = m % 7;
  const float* xp = xwsi + ((size_t)(b * 64 + ci) * CHn + d) * 3136;
  float s = 0.f;
#pragma unroll
  for (int yy = 0; yy < 8; ++yy)
#pragma unroll
    for (int xx = 0; xx < 8; ++xx)
      s += xp[(kh * 8 + yy) * 56 + (kw * 8 + xx)];
  P[((size_t)b * Mm + m) * 64 + ci] = s * (1.0f / 64.0f);
}

// xw1t[b][m][co] = b_wsi[co] + sum_ci W_wsi[co,ci]*P[b,m,ci]   (m padded to 608)
__global__ void k_build_xw1t(const float* __restrict__ P, const float* __restrict__ Wwsi,
                             const float* __restrict__ bwsi, bf16* __restrict__ xw1t) {
  int co = threadIdx.x;           // 64
  int m = blockIdx.x;             // 0..607
  int b = blockIdx.y;
  float v = 0.f;
  if (m < Mm) {
    v = bwsi[co];
    const float* pp = P + ((size_t)b * Mm + m) * 64;
#pragma unroll 4
    for (int ci = 0; ci < 64; ++ci) v += Wwsi[co * 64 + ci] * pp[ci];
  }
  xw1t[((size_t)b * Mp + m) * 64 + co] = (bf16)v;
}

// ---------------------------------------------------------------------------
// GEMM 1: x1t(b,N,64) = W_ct(64x64) * x_CT + b_ct     (K=64 -> 2 ksteps)
// ---------------------------------------------------------------------------
__global__ void k_gemm_x1(const bf16* __restrict__ xct_t, const bf16* __restrict__ wct,
                          const float* __restrict__ bias, bf16* __restrict__ x1t) {
  int lane = threadIdx.x, mtile = threadIdx.y, b = blockIdx.y;
  int n = blockIdx.x * 16 + (lane & 15);
  const bf16* colp = xct_t + ((size_t)b * Np + n) * 64;
  v16bf b0 = load_b_frag(colp, 0, lane);
  v16bf b1 = load_b_frag(colp, 1, lane);
  const bf16* arow = wct + (size_t)(mtile * 16 + (lane & 15)) * 64;
  v8f acc = {};
  acc = wmma_bf16(load_a_frag(arow, 0, lane), b0, acc);
  acc = wmma_bf16(load_a_frag(arow, 1, lane), b1, acc);
  int cbase = mtile * 16 + ((lane & 16) ? 8 : 0);
  v8bf o;
#pragma unroll
  for (int r = 0; r < 8; ++r) o[r] = (bf16)(acc[r] + bias[cbase + r]);
  *(v8bf*)(x1t + ((size_t)b * Np + n) * 64 + cbase) = o;
}

// ---------------------------------------------------------------------------
// GEMM 2: feature(b,N,608) = xw1^T(608x64) * x1(64xN)
// ---------------------------------------------------------------------------
__global__ void k_gemm_feature(const bf16* __restrict__ x1t, const bf16* __restrict__ xw1t,
                               bf16* __restrict__ feat) {
  int lane = threadIdx.x, w = threadIdx.y, b = blockIdx.y;
  int n = blockIdx.x * 16 + (lane & 15);
  const bf16* colp = x1t + ((size_t)b * Np + n) * 64;
  v16bf b0 = load_b_frag(colp, 0, lane);
  v16bf b1 = load_b_frag(colp, 1, lane);
  const bf16* Abase = xw1t + (size_t)b * Mp * 64;
  bf16* outp = feat + ((size_t)b * Np + n) * Mp;
  for (int mt = w; mt < Mp / 16; mt += 4) {
    const bf16* arow = Abase + (size_t)(mt * 16 + (lane & 15)) * 64;
    v8f acc = {};
    acc = wmma_bf16(load_a_frag(arow, 0, lane), b0, acc);
    acc = wmma_bf16(load_a_frag(arow, 1, lane), b1, acc);
    int mbase = mt * 16 + ((lane & 16) ? 8 : 0);
    v8bf o;
#pragma unroll
    for (int r = 0; r < 8; ++r) o[r] = (bf16)acc[r];
    *(v8bf*)(outp + mbase) = o;
  }
}

// ---------------------------------------------------------------------------
// GEMM 3: Att = sigmoid(conv3d(feature, W_att, pad=2)) as 125 shifted GEMMs.
// grid (Tt*Hh, Bn), block (32,4): wave = x-tile of 16.
// W_att tap-slab (16x608 bf16) is double-buffered in LDS, staged with the
// gfx1250 async global->LDS copy and consumed via ds loads; B streams from L2.
// ---------------------------------------------------------------------------
__global__ void __launch_bounds__(128) k_gemm_att(const bf16* __restrict__ feat,
                                                  const bf16* __restrict__ watt,
                                                  float* __restrict__ Att) {
  __shared__ bf16 slab[2][16 * Mp];   // 2 x 19456 B
  int lane = threadIdx.x, xt = threadIdx.y, b = blockIdx.y;
  int tid = xt * 32 + lane;
  int z = blockIdx.x / Hh, y = blockIdx.x % Hh;
  int xl = xt * 16 + (lane & 15);

  int dz0 = (2 - z) > 0 ? (2 - z) : 0;
  int dz1 = (Tt + 2 - z) < 5 ? (Tt + 2 - z) : 5;
  int dy0 = (2 - y) > 0 ? (2 - y) : 0;
  int dy1 = (Hh + 2 - y) < 5 ? (Hh + 2 - y) : 5;
  int nyy = dy1 - dy0;
  int ntap = (dz1 - dz0) * nyy * 5;

  // i -> tap index (uniform across the workgroup)
  auto tap_of = [&](int i) {
    int dz = dz0 + i / (nyy * 5);
    int r = i % (nyy * 5);
    int dy = dy0 + r / 5;
    int dx = r % 5;
    return (dz * 5 + dy) * 5 + dx;
  };

  stage_slab(watt + (size_t)tap_of(0) * 16 * Mp, &slab[0][0], tid);
  async_wait_all();
  __syncthreads();

  v8f acc = {};
  const bf16* fb = feat + (size_t)b * Np * Mp;
  int p = 0;
  for (int i = 0; i < ntap; ++i) {
    if (i + 1 < ntap) {
      // async-prefetch next tap's A slab into the other LDS buffer
      stage_slab(watt + (size_t)tap_of(i + 1) * 16 * Mp, &slab[p ^ 1][0], tid);
      // and hint the next tap's feature row toward the caches
      int tn = tap_of(i + 1);
      int dzn = tn / 25, dyn = (tn / 5) % 5, dxn = tn % 5;
      int xxn = xl + dxn - 2;
      int xcn = xxn < 0 ? 0 : (xxn > Ww - 1 ? Ww - 1 : xxn);
      __builtin_prefetch(
          fb + (size_t)(((z + dzn - 2) * Hh + (y + dyn - 2)) * Ww + xcn) * Mp, 0, 0);
    }
    int tap = tap_of(i);
    int dz = tap / 25, dy = (tap / 5) % 5, dx = tap % 5;
    int zz = z + dz - 2, yy = y + dy - 2;
    int xx = xl + dx - 2;
    bool xok = (xx >= 0) && (xx < Ww);
    int xc = xx < 0 ? 0 : (xx > Ww - 1 ? Ww - 1 : xx);
    const bf16* colp = fb + (size_t)((zz * Hh + yy) * Ww + xc) * Mp;
    const bf16* arow = &slab[p][(lane & 15) * Mp];
    for (int ks = 0; ks < 19; ++ks) {
      v16bf bfrag = load_b_frag(colp, ks, lane);
      if (!xok) bfrag = zero16();
      acc = wmma_bf16(load_a_frag(arow, ks, lane), bfrag, acc);
    }
    async_wait_all();
    __syncthreads();
    p ^= 1;
  }

  if (xl < Ww) {
    int n = (z * Hh + y) * Ww + xl;
#pragma unroll
    for (int r = 0; r < 8; ++r) {
      int oc = r + ((lane & 16) ? 8 : 0);
      if (oc < 13)
        Att[((size_t)(b * 13 + oc)) * Np + n] = 1.0f / (1.0f + __expf(-acc[r]));
    }
  }
}

// ---------------------------------------------------------------------------
// GEMM 4: kern(b,49,N) = W_inv(49x588) * feature + b_inv ; block (32,4) wave=mtile
// ---------------------------------------------------------------------------
__global__ void k_gemm_kern(const bf16* __restrict__ feat, const bf16* __restrict__ winv,
                            const float* __restrict__ binv, float* __restrict__ kern) {
  int lane = threadIdx.x, mt = threadIdx.y, b = blockIdx.y;
  int n = blockIdx.x * 16 + (lane & 15);
  const bf16* colp = feat + ((size_t)b * Np + n) * Mp;
  const bf16* arow = winv + (size_t)(mt * 16 + (lane & 15)) * Mp;
  v8f acc = {};
  for (int ks = 0; ks < 19; ++ks)
    acc = wmma_bf16(load_a_frag(arow, ks, lane), load_b_frag(colp, ks, lane), acc);
#pragma unroll
  for (int r = 0; r < 8; ++r) {
    int kk = mt * 16 + r + ((lane & 16) ? 8 : 0);
    if (kk < 49) kern[((size_t)(b * 49 + kk)) * Np + n] = acc[r] + binv[kk];
  }
}

// ---------------------------------------------------------------------------
// 3x3x3 conv: xw3(b,16,12,56,56) = conv3d(x_WSI, W_rr, pad=1)
// ---------------------------------------------------------------------------
__global__ void k_conv_rr(const float* __restrict__ xwsi, const float* __restrict__ wrr,
                          float* __restrict__ xw3) {
  int pos = blockIdx.x * 256 + threadIdx.x;
  if (pos >= 3136) return;
  int y = pos / 56, x = pos % 56;
  int co = blockIdx.y / CHn, d = blockIdx.y % CHn, b = blockIdx.z;
  float acc = 0.f;
  for (int ci = 0; ci < 64; ++ci) {
    const float* xp = xwsi + ((size_t)(b * 64 + ci) * CHn) * 3136;
    const float* wp = wrr + (co * 64 + ci) * 27;
#pragma unroll
    for (int dz = 0; dz < 3; ++dz) {
      int zz = d + dz - 1;
      if (zz < 0 || zz >= CHn) continue;
#pragma unroll
      for (int dyy = 0; dyy < 3; ++dyy) {
        int yy = y + dyy - 1;
        if (yy < 0 || yy >= 56) continue;
#pragma unroll
        for (int dxx = 0; dxx < 3; ++dxx) {
          int xx = x + dxx - 1;
          if (xx < 0 || xx >= 56) continue;
          acc += wp[(dz * 3 + dyy) * 3 + dxx] * xp[(size_t)zz * 3136 + yy * 56 + xx];
        }
      }
    }
  }
  xw3[((size_t)(b * CRr + co) * CHn + d) * 3136 + pos] = acc;
}

// ---------------------------------------------------------------------------
// x_Att = x_FS * (1 + Att_FS), stored (b,N,256) bf16
// ---------------------------------------------------------------------------
__global__ void k_compose_xatt(const float* __restrict__ xct, const float* __restrict__ xw3,
                               const float* __restrict__ Att, bf16* __restrict__ xatt) {
  long long id = (long long)blockIdx.x * 256 + threadIdx.x;
  if (id >= (long long)Bn * Np * Cf) return;
  int ch = (int)(id % Cf);
  long long nn = id / Cf;
  int n = (int)(nn % Np);
  int b = (int)(nn / Np);
  float xfs, att;
  if (ch < 64) {
    xfs = xct[((size_t)(b * 64 + ch)) * Np + n];
    att = Att[(size_t)(b * 13) * Np + n];
  } else {
    int u = ch - 64;
    int pos = n % 3136;
    xfs = xw3[((size_t)(b * CRr + u / CHn) * CHn + (u % CHn)) * 3136 + pos];
    att = Att[((size_t)(b * 13 + 1 + u / CRr)) * Np + n];
  }
  xatt[id] = (bf16)(xfs * (1.0f + att));
}

// ---------------------------------------------------------------------------
// GEMM p1 / p2 (K=256 -> 8 ksteps)
// ---------------------------------------------------------------------------
__global__ void k_gemm_p1(const bf16* __restrict__ xatt, const bf16* __restrict__ wp1,
                          const float* __restrict__ bias, float* __restrict__ z) {
  int lane = threadIdx.x, w = threadIdx.y, b = blockIdx.y;
  int n = blockIdx.x * 16 + (lane & 15);
  const bf16* colp = xatt + ((size_t)b * Np + n) * Cf;
  v8f acc[4] = {{}, {}, {}, {}};
  for (int ks = 0; ks < 8; ++ks) {
    v16bf bfrag = load_b_frag(colp, ks, lane);
#pragma unroll
    for (int i = 0; i < 4; ++i) {
      int mt = w + i * 4;
      acc[i] = wmma_bf16(load_a_frag(wp1 + (size_t)(mt * 16 + (lane & 15)) * Cf, ks, lane),
                         bfrag, acc[i]);
    }
  }
  float* outp = z + ((size_t)b * Np + n) * Cf;
#pragma unroll
  for (int i = 0; i < 4; ++i) {
    int cbase = (w + i * 4) * 16 + ((lane & 16) ? 8 : 0);
#pragma unroll
    for (int r = 0; r < 8; ++r) outp[cbase + r] = acc[i][r] + bias[cbase + r];
  }
}

__global__ void k_gemm_p2(const bf16* __restrict__ fB, const bf16* __restrict__ wp2,
                          const float* __restrict__ bias, float* __restrict__ out) {
  int lane = threadIdx.x, mt = threadIdx.y, b = blockIdx.y;
  int n = blockIdx.x * 16 + (lane & 15);
  const bf16* colp = fB + ((size_t)b * Np + n) * Cf;
  const bf16* arow = wp2 + (size_t)(mt * 16 + (lane & 15)) * Cf;
  v8f acc = {};
  for (int ks = 0; ks < 8; ++ks)
    acc = wmma_bf16(load_a_frag(arow, ks, lane), load_b_frag(colp, ks, lane), acc);
#pragma unroll
  for (int r = 0; r < 8; ++r) {
    int c = mt * 16 + r + ((lane & 16) ? 8 : 0);
    out[((size_t)(b * 64 + c)) * Np + n] = acc[r] + bias[c];
  }
}

// ---------------------------------------------------------------------------
// BatchNorm stats over (b,n) rows of an (b,N,256) fp32 buffer
// ---------------------------------------------------------------------------
__global__ void k_zero_stats(float* st) { st[threadIdx.x] = 0.f; st[256 + threadIdx.x] = 0.f; }

__global__ void k_stats(const float* __restrict__ z, float* __restrict__ st) {
  int ch = threadIdx.x;
  long long rows = (long long)Bn * Np;
  long long per = (rows + gridDim.x - 1) / gridDim.x;
  long long r0 = (long long)blockIdx.x * per;
  long long r1 = r0 + per; if (r1 > rows) r1 = rows;
  float s = 0.f, s2 = 0.f;
  for (long long r = r0; r < r1; ++r) {
    float v = z[r * Cf + ch];
    s += v; s2 += v * v;
  }
  atomicAdd(&st[ch], s);
  atomicAdd(&st[256 + ch], s2);
}

__global__ void k_finalize(const float* __restrict__ st, const float* __restrict__ g,
                           const float* __restrict__ be, float* __restrict__ scsh) {
  int ch = threadIdx.x;
  float inv = 1.0f / (float)((long long)Bn * Np);
  float mean = st[ch] * inv;
  float var = st[256 + ch] * inv - mean * mean;
  float sc = g[ch] * rsqrtf(var + 1e-5f);
  scsh[ch] = sc;
  scsh[256 + ch] = be[ch] - mean * sc;
}

__global__ void k_norm_relu(const float* __restrict__ z, const float* __restrict__ scsh,
                            bf16* __restrict__ o) {
  long long id = (long long)blockIdx.x * 256 + threadIdx.x;
  if (id >= (long long)Bn * Np * Cf) return;
  int ch = (int)(id % Cf);
  float v = z[id] * scsh[ch] + scsh[256 + ch];
  o[id] = (bf16)(v > 0.f ? v : 0.f);
}

// ---------------------------------------------------------------------------
// Involution: inv[b,n,ch] = sum_{i,j} kern[b,ij,n] * x1[b, (z,y+i-3,x+j-3), ch]
// ---------------------------------------------------------------------------
__global__ void k_involution(const bf16* __restrict__ x1, const float* __restrict__ kern,
                             float* __restrict__ inv) {
  int ch = threadIdx.x;
  int pos = blockIdx.x, z = blockIdx.y, b = blockIdx.z;
  int y = pos / 56, x = pos % 56;
  int n = z * 3136 + pos;
  float acc = 0.f;
  for (int i = 0; i < 7; ++i) {
    int yy = y + i - 3;
    if (yy < 0 || yy >= 56) continue;
    for (int j = 0; j < 7; ++j) {
      int xx = x + j - 3;
      if (xx < 0 || xx >= 56) continue;
      float kv = kern[((size_t)(b * 49 + i * 7 + j)) * Np + n];
      float av = (float)x1[((size_t)b * Np + (size_t)z * 3136 + yy * 56 + xx) * Cf + ch];
      acc += kv * av;
    }
  }
  inv[((size_t)b * Np + n) * Cf + ch] = acc;
}

__global__ void k_compose_final(const float* __restrict__ inv, const float* __restrict__ scsh,
                                const bf16* __restrict__ xatt, bf16* __restrict__ fB) {
  long long id = (long long)blockIdx.x * 256 + threadIdx.x;
  if (id >= (long long)Bn * Np * Cf) return;
  int ch = (int)(id % Cf);
  fB[id] = (bf16)(inv[id] * scsh[ch] + scsh[256 + ch] + (float)xatt[id]);
}

// ---------------------------------------------------------------------------
extern "C" void kernel_launch(void* const* d_in, const int* in_sizes, int n_in,
                              void* d_out, int out_size, void* d_ws, size_t ws_size,
                              hipStream_t stream) {
  const float* x_CT  = (const float*)d_in[0];
  const float* x_WSI = (const float*)d_in[1];
  const float* W_ct  = (const float*)d_in[2];
  const float* b_ct  = (const float*)d_in[3];
  const float* W_wsi = (const float*)d_in[4];
  const float* b_wsi = (const float*)d_in[5];
  const float* W_att = (const float*)d_in[6];
  const float* W_inv = (const float*)d_in[7];
  const float* b_inv = (const float*)d_in[8];
  const float* W_rr  = (const float*)d_in[9];
  const float* g1    = (const float*)d_in[10];
  const float* be1   = (const float*)d_in[11];
  const float* g2    = (const float*)d_in[12];
  const float* be2   = (const float*)d_in[13];
  const float* W_p1  = (const float*)d_in[14];
  const float* b_p1  = (const float*)d_in[15];
  const float* W_p2  = (const float*)d_in[16];
  const float* b_p2  = (const float*)d_in[17];
  float* out = (float*)d_out;

  char* base = (char*)d_ws;
  size_t cur = 0;
  auto alloc = [&](size_t bytes) -> void* {
    void* p = base + cur;
    cur = (cur + bytes + 255) & ~(size_t)255;
    return p;
  };

  bf16* xct_t   = (bf16*)alloc((size_t)Bn * Np * 64 * 2);
  bf16* x1t     = (bf16*)alloc((size_t)Bn * Np * 64 * 2);
  bf16* wct_bf  = (bf16*)alloc(64 * 64 * 2);
  bf16* wp1_bf  = (bf16*)alloc((size_t)Cf * Cf * 2);
  bf16* wp2_bf  = (bf16*)alloc(64 * Cf * 2);
  bf16* winv_bf = (bf16*)alloc(64 * Mp * 2);
  bf16* watt_bf = (bf16*)alloc((size_t)125 * 16 * Mp * 2);
  float* Pbuf   = (float*)alloc((size_t)Bn * Mm * 64 * 4);
  bf16* xw1t    = (bf16*)alloc((size_t)Bn * Mp * 64 * 2);
  bf16* feat    = (bf16*)alloc((size_t)Bn * Np * Mp * 2);   // reused as z (fp32)
  float* zbuf   = (float*)feat;                             // Bn*Np*Cf*4 <= feat bytes
  float* Att    = (float*)alloc((size_t)Bn * 13 * Np * 4);
  float* xw3    = (float*)alloc((size_t)Bn * CRr * CHn * 3136 * 4);
  float* kern   = (float*)alloc((size_t)Bn * 49 * Np * 4);
  bf16* xatt_t  = (bf16*)alloc((size_t)Bn * Np * Cf * 2);
  bf16* xatt1_t = (bf16*)alloc((size_t)Bn * Np * Cf * 2);   // reused as fB
  float* st     = (float*)alloc(512 * 4);
  float* scsh   = (float*)alloc(512 * 4);
  (void)ws_size; (void)in_sizes; (void)n_in; (void)out_size;

  dim3 w4(32, 4);

  // --- weight / input repack --------------------------------------------
  k_cvt_bf16<<<(64 * 64 + 255) / 256, 256, 0, stream>>>(W_ct, wct_bf, 64 * 64);
  k_cvt_bf16<<<(Cf * Cf + 255) / 256, 256, 0, stream>>>(W_p1, wp1_bf, Cf * Cf);
  k_cvt_bf16<<<(64 * Cf + 255) / 256, 256, 0, stream>>>(W_p2, wp2_bf, 64 * Cf);
  k_pack_winv<<<(64 * Mp + 255) / 256, 256, 0, stream>>>(W_inv, winv_bf);
  k_pack_watt<<<(int)(((size_t)125 * 16 * Mp + 255) / 256), 256, 0, stream>>>(W_att, watt_bf);
  k_transpose_xct<<<dim3(Np / 256, Bn), 256, 0, stream>>>(x_CT, xct_t);
  k_pool_wsi<<<dim3(Mm, Bn), 64, 0, stream>>>(x_WSI, Pbuf);
  k_build_xw1t<<<dim3(Mp, Bn), 64, 0, stream>>>(Pbuf, W_wsi, b_wsi, xw1t);

  // --- main WMMA pipeline -----------------------------------------------
  k_gemm_x1<<<dim3(NT, Bn), w4, 0, stream>>>(xct_t, wct_bf, b_ct, x1t);
  k_gemm_feature<<<dim3(NT, Bn), w4, 0, stream>>>(x1t, xw1t, feat);
  k_gemm_att<<<dim3(Tt * Hh, Bn), w4, 0, stream>>>(feat, watt_bf, Att);
  k_gemm_kern<<<dim3(NT, Bn), w4, 0, stream>>>(feat, winv_bf, b_inv, kern);
  k_conv_rr<<<dim3(13, CRr * CHn, Bn), 256, 0, stream>>>(x_WSI, W_rr, xw3);
  k_compose_xatt<<<(int)(((size_t)Bn * Np * Cf + 255) / 256), 256, 0, stream>>>(x_CT, xw3, Att, xatt_t);

  // p1 + bnorm1 + relu (zbuf overwrites feat, which is no longer needed)
  k_gemm_p1<<<dim3(NT, Bn), w4, 0, stream>>>(xatt_t, wp1_bf, b_p1, zbuf);
  k_zero_stats<<<1, 256, 0, stream>>>(st);
  k_stats<<<196, 256, 0, stream>>>(zbuf, st);
  k_finalize<<<1, 256, 0, stream>>>(st, g1, be1, scsh);
  k_norm_relu<<<(int)(((size_t)Bn * Np * Cf + 255) / 256), 256, 0, stream>>>(zbuf, scsh, xatt1_t);

  // involution (writes back into zbuf), bnorm2, residual, p2
  k_involution<<<dim3(3136, Tt, Bn), 256, 0, stream>>>(xatt1_t, kern, zbuf);
  k_zero_stats<<<1, 256, 0, stream>>>(st);
  k_stats<<<196, 256, 0, stream>>>(zbuf, st);
  k_finalize<<<1, 256, 0, stream>>>(st, g2, be2, scsh);
  k_compose_final<<<(int)(((size_t)Bn * Np * Cf + 255) / 256), 256, 0, stream>>>(zbuf, scsh, xatt_t, xatt1_t);
  k_gemm_p2<<<dim3(NT, Bn), w4, 0, stream>>>(xatt1_t, wp2_bf, b_p2, out);
}
